// RelationshipDetector_31061203485304
// MI455X (gfx1250) — matrix operand order repaired and checked by reading, not compiled, not run on hardware
//
#include <hip/hip_runtime.h>

#define B_   1024
#define D_   1024
#define N_   51
#define H_   10
#define NIT  5

typedef unsigned int       u32;
typedef unsigned short     u16;
typedef unsigned long long u64;

typedef __attribute__((ext_vector_type(16))) __bf16 v16bf;
typedef __attribute__((ext_vector_type(8)))  float  v8f;
typedef __attribute__((ext_vector_type(4)))  unsigned int v4u;
typedef __attribute__((ext_vector_type(8)))  int    v8i;
typedef __attribute__((ext_vector_type(4)))  int    v4i;

#if defined(__has_builtin)
#if __has_builtin(__builtin_amdgcn_tensor_load_to_lds) && __has_builtin(__builtin_amdgcn_s_wait_tensorcnt)
#define USE_TDM 1
#endif
#endif
#ifndef USE_TDM
#define USE_TDM 0
#endif

union BFrag { v16bf v; u32 u[8]; };

__device__ __forceinline__ u16 f32_to_bf16(float f) {
  u32 u = __float_as_uint(f);
  u32 r = (u + 0x7FFFu + ((u >> 16) & 1u)) >> 16;
  return (u16)r;
}

// ---------------------------------------------------------------------------
__global__ __launch_bounds__(256) void zero_f32_kernel(float* p, int n) {
  int i = blockIdx.x * 256 + threadIdx.x;
  if (i < n) p[i] = 0.0f;
}
__global__ __launch_bounds__(256) void zero_u16_kernel(u16* p, int n) {
  int i = blockIdx.x * 256 + threadIdx.x;
  if (i < n) p[i] = 0;
}

// tiled transpose + f32->bf16:  src[R,C] f32 row-major  ->  dst[C,R] bf16
__global__ __launch_bounds__(256) void transpose_bf16_kernel(const float* __restrict__ src,
                                                             u16* __restrict__ dst,
                                                             int R, int C) {
  __shared__ u16 tile[32][33];
  int c0 = blockIdx.x * 32, r0 = blockIdx.y * 32;
  int tx = threadIdx.x & 31, ty = threadIdx.x >> 5;   // 32 x 8
  #pragma unroll
  for (int i = 0; i < 32; i += 8)
    tile[ty + i][tx] = f32_to_bf16(src[(size_t)(r0 + ty + i) * C + (c0 + tx)]);
  __syncthreads();
  #pragma unroll
  for (int i = 0; i < 32; i += 8)
    dst[(size_t)(c0 + ty + i) * R + (r0 + tx)] = tile[tx][ty + i];
}

// ---------------------------------------------------------------------------
// Tensor Data Mover: 2D tile (rows x 32 bf16) global -> LDS, with 1-dword pad
// per 16-dword (64B) row => LDS row stride 34 u16 (matches fragment layout).
#if USE_TDM
__device__ __forceinline__ void tdm_load_tile(u32 lds_off, const u16* gptr,
                                              u32 stride_elems, u32 rows) {
  u64 ga = (u64)(size_t)gptr;
  v4u g0;
  g0.x = 1u;                                        // count=1, user mode
  g0.y = lds_off;                                   // lds_addr (bytes)
  g0.z = (u32)ga;                                   // global_addr[31:0]
  g0.w = (u32)((ga >> 32) & 0x01FFFFFFu) | (2u << 30);  // addr[56:32] | type=2
  v8i g1;
  // data_size=2B(1<<16) | pad_enable(1<<20) | pad_interval=3 (16 dwords)<<22 | pad_amount=0 (1 dword)
  g1[0] = (int)((1u << 16) | (1u << 20) | (3u << 22));
  g1[1] = (int)(32u << 16);                         // tensor_dim0 = 32 (low 16 in [31:16])
  g1[2] = (int)((rows & 0xFFFFu) << 16);            // tensor_dim0 hi=0 | tensor_dim1 lo
  g1[3] = (int)(((rows >> 16) & 0xFFFFu) | (32u << 16)); // tensor_dim1 hi | tile_dim0=32
  g1[4] = (int)(rows & 0xFFFFu);                    // tile_dim1 = rows, tile_dim2 = 0
  g1[5] = (int)stride_elems;                        // tensor_dim0_stride[31:0]
  g1[6] = 0;                                        // stride hi | dim1_stride lo
  g1[7] = 0;
  v4i gz = {};
  v8i gz8 = {};
  __builtin_amdgcn_tensor_load_to_lds(g0, g1, gz, gz, gz8, 0);
}
#endif

// ---------------------------------------------------------------------------
// WMMA GEMM:  C[M,N] = A[M,K](bf16 row-major) * Bt[N,K](bf16 row-major)^T
// Block tile 128x64, 8 waves (wave32), wave = 32x32 via 2x2 v_wmma 16x16x32.
// Tiles staged via TDM (double-buffered) when available.
enum { EP_BN = 0, EP_BIAS_F32 = 1 };

template <int EP>
__global__ __launch_bounds__(256)
void gemm_wmma_kernel(const u16* __restrict__ A, const u16* __restrict__ Bt,
                      int M, int N, int K,
                      const float* __restrict__ bias,
                      const float* __restrict__ gg, const float* __restrict__ bb,
                      const float* __restrict__ mm, const float* __restrict__ vv,
                      float* __restrict__ outF, u16* __restrict__ outB) {
  constexpr int AS = 34;                 // LDS row stride in u16 (64B + 4B pad)
  constexpr int BS = 34;
  __shared__ u16 As[2][128 * AS];
  __shared__ u16 Bs[2][64 * BS];

  const int tid  = threadIdx.x;
  const int wave = tid >> 5;
  const int lane = tid & 31;
  const int wm   = wave >> 1;            // 0..3 -> 32-row slab
  const int wn   = wave & 1;             // 0..1 -> 32-col slab
  const int half = lane >> 4;
  const int lm   = lane & 15;
  const int rowBase = blockIdx.y * 128;
  const int colBase = blockIdx.x * 64;

  const u16* Ag = A  + (size_t)rowBase * K;
  const u16* Bg = Bt + (size_t)colBase * K;

  v8f acc[2][2] = {};
  const int nsteps = K >> 5;

#if USE_TDM
  const u32 offA0 = (u32)(size_t)(void*)&As[0][0];
  const u32 offA1 = (u32)(size_t)(void*)&As[1][0];
  const u32 offB0 = (u32)(size_t)(void*)&Bs[0][0];
  const u32 offB1 = (u32)(size_t)(void*)&Bs[1][0];
  if (wave == 0) {
    tdm_load_tile(offA0, Ag, (u32)K, 128u);
    tdm_load_tile(offB0, Bg, (u32)K, 64u);
  }
#endif

  for (int s = 0; s < nsteps; ++s) {
    const int cur = s & 1;
#if USE_TDM
    if (wave == 0) {
      __builtin_amdgcn_s_wait_tensorcnt(0);      // tiles for step s landed
      if (s + 1 < nsteps) {                       // prefetch step s+1
        tdm_load_tile(cur ? offA0 : offA1, Ag + (size_t)(s + 1) * 32, (u32)K, 128u);
        tdm_load_tile(cur ? offB0 : offB1, Bg + (size_t)(s + 1) * 32, (u32)K, 64u);
      }
    }
    __syncthreads();                              // buffer `cur` visible to all
#else
    __syncthreads();
    {   // stage A: 128x32 bf16 (2048 dwords / 256 threads)
      const int k0 = s * 32;
      const u32* Agd = (const u32*)(Ag + k0);
      const int Kd2 = K >> 1;
      #pragma unroll
      for (int i = 0; i < 8; ++i) {
        int idx = tid + i * 256;
        int r = idx >> 4, cp = idx & 15;
        u32 d = Agd[(size_t)r * Kd2 + cp];
        *(u32*)&As[cur][r * AS + cp * 2] = d;
      }
      // stage B: 64 rows x 32 bf16 from Bt (coalesced along K)
      #pragma unroll
      for (int i = 0; i < 4; ++i) {
        int idx = tid + i * 256;                  // 0..1023 dwords
        int nn = idx >> 3, cp = idx & 7;
        u32 d = ((const u32*)(Bg + (size_t)nn * K + k0))[cp];
        *(u32*)&Bs[cur][nn * BS + cp * 2] = d;
      }
    }
    __syncthreads();
#endif

    // ---- fragments per ISA 7.12.2 layouts
    BFrag a[2], b[2];
    #pragma unroll
    for (int tr = 0; tr < 2; ++tr) {
      const u16* base = &As[cur][(wm * 32 + tr * 16 + lm) * AS];
      #pragma unroll
      for (int j = 0; j < 8; ++j) {
        int g = j >> 2, jj = j & 3;
        int k = 16 * g + 8 * half + 2 * jj;
        a[tr].u[j] = *(const u32*)&base[k];
      }
    }
    #pragma unroll
    for (int tc = 0; tc < 2; ++tc) {
      const u16* base = &Bs[cur][(wn * 32 + tc * 16 + lm) * BS];
      #pragma unroll
      for (int j = 0; j < 8; ++j) {
        int k = 16 * (j >> 2) + 8 * half + 2 * (j & 3);
        b[tc].u[j] = *(const u32*)&base[k];
      }
    }
    #pragma unroll
    for (int tr = 0; tr < 2; ++tr)
      #pragma unroll
      for (int tc = 0; tc < 2; ++tc)
        acc[tr][tc] = __builtin_amdgcn_wmma_f32_16x16x32_bf16(
            false, a[tr].v, false, b[tc].v, (short)0, acc[tr][tc], false, false);

    __syncthreads();                              // reads done; buffer reusable
  }

  // ---- epilogue (C/D: VGPR i -> M = i + 8*half, N = lm)
  #pragma unroll
  for (int tr = 0; tr < 2; ++tr) {
    #pragma unroll
    for (int tc = 0; tc < 2; ++tc) {
      int colg = colBase + wn * 32 + tc * 16 + lm;
      #pragma unroll
      for (int i = 0; i < 8; ++i) {
        int rowg = rowBase + wm * 32 + tr * 16 + i + 8 * half;
        float val = acc[tr][tc][i];
        if constexpr (EP == EP_BN) {
          val += bias[colg];
          val = fmaxf(val, 0.0f);
          val = (val - mm[colg]) * rsqrtf(vv[colg] + 1e-3f) * gg[colg] + bb[colg];
          outB[(size_t)rowg * N + colg] = f32_to_bf16(val);
        } else {
          outF[(size_t)rowg * N + colg] = val + bias[colg];
        }
      }
    }
  }
}

// ---------------------------------------------------------------------------
// vis_vec = mean over 100 spatial positions (float4 / b128 loads)
__global__ __launch_bounds__(256) void mean_kernel(const float* __restrict__ fm,
                                                   float* __restrict__ vis) {
  int i = blockIdx.x * 256 + threadIdx.x;         // over B*D/4
  if (i >= B_ * D_ / 4) return;
  int b = i >> 8, d4 = i & 255;
  const float4* f = (const float4*)fm + (size_t)b * 100 * 256 + d4;
  float ax = 0, ay = 0, az = 0, aw = 0;
  for (int p = 0; p < 100; ++p) {
    float4 t = f[(size_t)p * 256];
    ax += t.x; ay += t.y; az += t.z; aw += t.w;
  }
  float4 r; r.x = ax * 0.01f; r.y = ay * 0.01f; r.z = az * 0.01f; r.w = aw * 0.01f;
  ((float4*)vis)[i] = r;
}

// vis_vec = sum(featuremap * attn) (float4 / b128 loads)
__global__ __launch_bounds__(256) void attnsum_kernel(const float* __restrict__ fm,
                                                      const float* __restrict__ attn,
                                                      float* __restrict__ vis) {
  int i = blockIdx.x * 256 + threadIdx.x;
  if (i >= B_ * D_ / 4) return;
  int b = i >> 8, d4 = i & 255;
  const float4* f = (const float4*)fm + (size_t)b * 100 * 256 + d4;
  const float* a = attn + (size_t)b * 100;
  float ax = 0, ay = 0, az = 0, aw = 0;
  for (int p = 0; p < 100; ++p) {
    float4 t = f[(size_t)p * 256];
    float w = a[p];
    ax += t.x * w; ay += t.y * w; az += t.z * w; aw += t.w * w;
  }
  float4 r; r.x = ax; r.y = ay; r.z = az; r.w = aw;
  ((float4*)vis)[i] = r;
}

// combined = concat(nonvis, vis) in f32 and bf16
__global__ __launch_bounds__(256) void combined_kernel(const float* __restrict__ nonvis,
                                                       const float* __restrict__ vis,
                                                       float* __restrict__ cf,
                                                       u16* __restrict__ cb) {
  int i = blockIdx.x * 256 + threadIdx.x;
  if (i >= B_ * 3072) return;
  int b = i / 3072;
  int j = i - b * 3072;
  float v = (j < 2048) ? nonvis[(size_t)b * 2048 + j]
                       : vis[(size_t)b * 1024 + (j - 2048)];
  cf[i] = v;
  cb[i] = f32_to_bf16(v);
}

// score += combined @ Ws + bs
__global__ __launch_bounds__(64) void score_kernel(const float* __restrict__ cf,
                                                   const float* __restrict__ Ws,
                                                   const float* __restrict__ bs,
                                                   float* __restrict__ score) {
  int b = blockIdx.x;
  int n = threadIdx.x;
  if (n >= N_) return;
  const float* row = cf + (size_t)b * 3072;
  float acc = bs[n];
  for (int k = 0; k < 3072; ++k) acc += row[k] * Ws[(size_t)k * N_ + n];
  score[(size_t)b * N_ + n] += acc;
}

// GRU pointwise fusion
__global__ __launch_bounds__(256) void gru_kernel(const float* __restrict__ mx,
                                                  const float* __restrict__ mi,
                                                  float* __restrict__ h,
                                                  u16* __restrict__ hb) {
  int i = blockIdx.x * 256 + threadIdx.x;
  if (i >= B_ * D_) return;
  int b = i >> 10, d = i & 1023;
  const float* mxr = mx + (size_t)b * 3072;
  const float* mir = mi + (size_t)b * 3072;
  float xz = mxr[d], xr = mxr[1024 + d], xh = mxr[2048 + d];
  float iz = mir[d], ir = mir[1024 + d], ih = mir[2048 + d];
  float zg = 1.0f / (1.0f + __expf(-(xz + iz)));
  float rg = 1.0f / (1.0f + __expf(-(xr + ir)));
  float hh = tanhf(xh + rg * ih);
  float hn = zg * h[i] + (1.0f - zg) * hh;
  h[i] = hn;
  hb[i] = f32_to_bf16(hn);
}

// transposed conv 3x3 stride 2 SAME: out[yo] gathers in[yi], yo = 2*yi + ky - 1
template <int S, int Ci, int Co, bool RELU>
__global__ __launch_bounds__(256) void convt_kernel(const float* __restrict__ in,
                                                    const float* __restrict__ w,
                                                    const float* __restrict__ bias,
                                                    float* __restrict__ out) {
  const int SO = 2 * S;
  int idx = blockIdx.x * 256 + threadIdx.x;
  if (idx >= B_ * SO * SO * Co) return;
  int co = idx % Co; int t = idx / Co;
  int xo = t % SO;   t /= SO;
  int yo = t % SO;   int b = t / SO;
  float acc = bias[co];
  for (int ky = 0; ky < 3; ++ky) {
    int yn = yo + 1 - ky;
    if (yn < 0 || (yn & 1)) continue;
    int yi = yn >> 1; if (yi >= S) continue;
    for (int kx = 0; kx < 3; ++kx) {
      int xn = xo + 1 - kx;
      if (xn < 0 || (xn & 1)) continue;
      int xi = xn >> 1; if (xi >= S) continue;
      const float* ip = in + ((size_t)(b * S + yi) * S + xi) * Ci;
      const float* wp = w + (size_t)(ky * 3 + kx) * Ci * Co + co;
      for (int ci = 0; ci < Ci; ++ci) acc += ip[ci] * wp[(size_t)ci * Co];
    }
  }
  out[idx] = RELU ? fmaxf(acc, 0.0f) : acc;
}

// bilinear resize (B,16,16) -> (B,10,10), half-pixel centers
__global__ __launch_bounds__(256) void resize_kernel(const float* __restrict__ a4,
                                                     float* __restrict__ attn) {
  int idx = blockIdx.x * 256 + threadIdx.x;
  if (idx >= B_ * 100) return;
  int p = idx % 100; int b = idx / 100;
  int y = p / 10, x = p % 10;
  float sy = (y + 0.5f) * 1.6f - 0.5f;
  float sx = (x + 0.5f) * 1.6f - 0.5f;
  int y0 = (int)floorf(sy), x0 = (int)floorf(sx);
  float fy = sy - (float)y0, fx = sx - (float)x0;
  int y1 = y0 + 1 < 15 ? y0 + 1 : 15;
  int x1 = x0 + 1 < 15 ? x0 + 1 : 15;
  const float* q = a4 + (size_t)b * 256;
  float v00 = q[y0 * 16 + x0], v01 = q[y0 * 16 + x1];
  float v10 = q[y1 * 16 + x0], v11 = q[y1 * 16 + x1];
  attn[idx] = v00 * (1.f - fy) * (1.f - fx) + v01 * (1.f - fy) * fx +
              v10 * fy * (1.f - fx) + v11 * fy * fx;
}

// softmax over 100 per batch row (in place)
__global__ __launch_bounds__(128) void softmax_kernel(float* __restrict__ attn) {
  __shared__ float red[128];
  int b = blockIdx.x, t = threadIdx.x;
  float* row = attn + (size_t)b * 100;
  float v = (t < 100) ? row[t] : -1e30f;
  red[t] = v; __syncthreads();
  for (int s = 64; s > 0; s >>= 1) { if (t < s) red[t] = fmaxf(red[t], red[t + s]); __syncthreads(); }
  float mx = red[0]; __syncthreads();
  float e = (t < 100) ? __expf(v - mx) : 0.0f;
  red[t] = e; __syncthreads();
  for (int s = 64; s > 0; s >>= 1) { if (t < s) red[t] += red[t + s]; __syncthreads(); }
  float inv = 1.0f / red[0];
  if (t < 100) row[t] = e * inv;
}

__global__ __launch_bounds__(256) void final_kernel(const float* __restrict__ score,
                                                    const float* __restrict__ prior,
                                                    float* __restrict__ out) {
  int i = blockIdx.x * 256 + threadIdx.x;
  if (i >= B_ * N_) return;
  out[i] = score[i] * (1.0f / 6.0f) + prior[i];
}

// ---------------------------------------------------------------------------
extern "C" void kernel_launch(void* const* d_in, const int* in_sizes, int n_in,
                              void* d_out, int out_size, void* d_ws, size_t ws_size,
                              hipStream_t stream) {
  const float* nonvis = (const float*)d_in[0];
  const float* fm     = (const float*)d_in[1];
  const float* prior  = (const float*)d_in[2];
  const float* W1     = (const float*)d_in[3];
  const float* b1     = (const float*)d_in[4];
  const float* g1     = (const float*)d_in[5];
  const float* be1    = (const float*)d_in[6];
  const float* m1     = (const float*)d_in[7];
  const float* v1     = (const float*)d_in[8];
  const float* W2     = (const float*)d_in[9];
  const float* b2     = (const float*)d_in[10];
  const float* g2     = (const float*)d_in[11];
  const float* be2    = (const float*)d_in[12];
  const float* m2     = (const float*)d_in[13];
  const float* v2     = (const float*)d_in[14];
  const float* Wk     = (const float*)d_in[15];
  const float* Uk     = (const float*)d_in[16];
  const float* bi     = (const float*)d_in[17];
  const float* br     = (const float*)d_in[18];
  const float* ct1w   = (const float*)d_in[19];
  const float* ct1b   = (const float*)d_in[20];
  const float* ct2w   = (const float*)d_in[21];
  const float* ct2b   = (const float*)d_in[22];
  const float* ct3w   = (const float*)d_in[23];
  const float* ct3b   = (const float*)d_in[24];
  const float* ct4w   = (const float*)d_in[25];
  const float* ct4b   = (const float*)d_in[26];
  const float* Ws     = (const float*)d_in[27];
  const float* bs     = (const float*)d_in[28];
  float* out = (float*)d_out;

  char* ws = (char*)d_ws;
  size_t off = 0;
  auto alloc = [&](size_t bytes) -> char* {
    char* p = ws + off;
    off += (bytes + 255) & ~(size_t)255;
    return p;
  };
  u16*   W1t    = (u16*)alloc((size_t)2048 * 3072 * 2);   // W1^T  (N x K)
  u16*   W2t    = (u16*)alloc((size_t)1024 * 2048 * 2);   // W2^T
  u16*   Wkt    = (u16*)alloc((size_t)3072 * 1024 * 2);   // Wk^T
  u16*   Ukt    = (u16*)alloc((size_t)3072 * 1024 * 2);   // Uk^T
  float* combF  = (float*)alloc((size_t)B_ * 3072 * 4);
  u16*   combB  = (u16*)alloc((size_t)B_ * 3072 * 2);
  u16*   x1b    = (u16*)alloc((size_t)B_ * 2048 * 2);
  u16*   x2b    = (u16*)alloc((size_t)B_ * 1024 * 2);
  float* hF     = (float*)alloc((size_t)B_ * 1024 * 4);
  u16*   hB     = (u16*)alloc((size_t)B_ * 1024 * 2);
  float* mxBuf  = (float*)alloc((size_t)B_ * 3072 * 4);
  float* miBuf  = (float*)alloc((size_t)B_ * 3072 * 4);
  float* score  = (float*)alloc((size_t)B_ * N_ * 4);
  float* a1     = (float*)alloc((size_t)B_ * 1024 * 4);
  float* a2     = (float*)alloc((size_t)B_ * 1024 * 4);
  float* a3     = (float*)alloc((size_t)B_ * 1024 * 4);
  float* a4     = (float*)alloc((size_t)B_ * 256 * 4);
  float* attn   = (float*)alloc((size_t)B_ * 100 * 4);
  float* vis    = (float*)alloc((size_t)B_ * 1024 * 4);

  // one-time weight transpose+convert (N x K bf16, TDM/coalesced-friendly)
  transpose_bf16_kernel<<<dim3(2048 / 32, 3072 / 32), 256, 0, stream>>>(W1, W1t, 3072, 2048);
  transpose_bf16_kernel<<<dim3(1024 / 32, 2048 / 32), 256, 0, stream>>>(W2, W2t, 2048, 1024);
  transpose_bf16_kernel<<<dim3(3072 / 32, 1024 / 32), 256, 0, stream>>>(Wk, Wkt, 1024, 3072);
  transpose_bf16_kernel<<<dim3(3072 / 32, 1024 / 32), 256, 0, stream>>>(Uk, Ukt, 1024, 3072);

  zero_f32_kernel<<<(B_ * 1024 + 255) / 256, 256, 0, stream>>>(hF, B_ * 1024);
  zero_u16_kernel<<<(B_ * 1024 + 255) / 256, 256, 0, stream>>>(hB, B_ * 1024);
  zero_f32_kernel<<<(B_ * N_ + 255) / 256, 256, 0, stream>>>(score, B_ * N_);

  mean_kernel<<<(B_ * D_ / 4 + 255) / 256, 256, 0, stream>>>(fm, vis);
  combined_kernel<<<(B_ * 3072 + 255) / 256, 256, 0, stream>>>(nonvis, vis, combF, combB);
  score_kernel<<<B_, 64, 0, stream>>>(combF, Ws, bs, score);

  for (int it = 0; it < NIT; ++it) {
    gemm_wmma_kernel<EP_BN><<<dim3(2048 / 64, B_ / 128), 256, 0, stream>>>(
        combB, W1t, B_, 2048, 3072, b1, g1, be1, m1, v1, nullptr, x1b);
    gemm_wmma_kernel<EP_BN><<<dim3(1024 / 64, B_ / 128), 256, 0, stream>>>(
        x1b, W2t, B_, 1024, 2048, b2, g2, be2, m2, v2, nullptr, x2b);
    gemm_wmma_kernel<EP_BIAS_F32><<<dim3(3072 / 64, B_ / 128), 256, 0, stream>>>(
        x2b, Wkt, B_, 3072, 1024, bi, nullptr, nullptr, nullptr, nullptr, mxBuf, nullptr);
    gemm_wmma_kernel<EP_BIAS_F32><<<dim3(3072 / 64, B_ / 128), 256, 0, stream>>>(
        hB, Ukt, B_, 3072, 1024, br, nullptr, nullptr, nullptr, nullptr, miBuf, nullptr);
    gru_kernel<<<(B_ * D_ + 255) / 256, 256, 0, stream>>>(mxBuf, miBuf, hF, hB);

    convt_kernel<1, 1024, 256, true><<<(B_ * 2 * 2 * 256 + 255) / 256, 256, 0, stream>>>(hF, ct1w, ct1b, a1);
    convt_kernel<2, 256, 64, true><<<(B_ * 4 * 4 * 64 + 255) / 256, 256, 0, stream>>>(a1, ct2w, ct2b, a2);
    convt_kernel<4, 64, 16, true><<<(B_ * 8 * 8 * 16 + 255) / 256, 256, 0, stream>>>(a2, ct3w, ct3b, a3);
    convt_kernel<8, 16, 1, false><<<(B_ * 16 * 16 + 255) / 256, 256, 0, stream>>>(a3, ct4w, ct4b, a4);
    resize_kernel<<<(B_ * 100 + 255) / 256, 256, 0, stream>>>(a4, attn);
    softmax_kernel<<<B_, 128, 0, stream>>>(attn);

    attnsum_kernel<<<(B_ * D_ / 4 + 255) / 256, 256, 0, stream>>>(fm, attn, vis);
    combined_kernel<<<(B_ * 3072 + 255) / 256, 256, 0, stream>>>(nonvis, vis, combF, combB);
    score_kernel<<<B_, 64, 0, stream>>>(combF, Ws, bs, score);
  }

  final_kernel<<<(B_ * N_ + 255) / 256, 256, 0, stream>>>(score, prior, out);
  (void)in_sizes; (void)n_in; (void)out_size; (void)ws_size;
}
